// LSTMModel_10995116278125
// MI455X (gfx1250) — compile-verified, matrix operand order checked by reading, not run on hardware
//
#include <hip/hip_runtime.h>
#include <hip/hip_bf16.h>

// Problem constants (from reference)
#define B_   128
#define T_   1024
#define F_   32
#define E_   24
#define U_   512
#define G4_  2048   // 4*U
#define NBLK 32
#define NTHR 256    // 8 wave32 per workgroup -> 256 waves total -> 4 tiles/wave

// LDS staging for the block's Wh slice: 64 columns x 512 K in bf16,
// row stride padded to 1040 B (260 dwords = 4 mod 64 banks -> conflict-free b128)
#define LDS_ROW_B 1040

typedef __bf16 v16bf __attribute__((ext_vector_type(16)));
typedef float  v8f   __attribute__((ext_vector_type(8)));

struct U4 { unsigned int x, y, z, w; };
struct U8 { U4 lo, hi; };   // 32 bytes == v16bf

static __device__ __forceinline__ float sigmoidf_(float x) {
  return 1.0f / (1.0f + __expf(-x));
}

// ---------------- one-time prep: transpose+convert weights to bf16 ----------------
// WhT[n*512+k] = bf16(Wh[k*2048+n])   (N-major so B-tile loads match WMMA layout)
// WxT[n*32+k]  = bf16(Wx[k*2048+n])   (K padded 24 -> 32 with zeros)
__global__ void prep_kernel(const float* __restrict__ Wh, const float* __restrict__ Wx,
                            __hip_bfloat16* __restrict__ WhT, __hip_bfloat16* __restrict__ WxT) {
  int idx = blockIdx.x * blockDim.x + threadIdx.x;
  const int nWh = G4_ * U_;
  const int nWx = G4_ * F_;
  if (idx < nWh) {
    int n = idx / U_, k = idx % U_;
    WhT[idx] = __float2bfloat16(Wh[(size_t)k * G4_ + n]);
  } else if (idx < nWh + nWx) {
    int j = idx - nWh;
    int n = j / F_, k = j % F_;
    float v = (k < E_) ? Wx[(size_t)k * G4_ + n] : 0.0f;
    WxT[j] = __float2bfloat16(v);
  }
}

// ---------------- embedding: x = sigmoid(data@W_emb + b_emb), bf16, K padded to 32 ----------------
// Xbf layout: [t][b][32] so Phase A can load A tiles with 16B chunks.
__global__ void embed_kernel(const float* __restrict__ data, const float* __restrict__ W_emb,
                             const float* __restrict__ b_emb, __hip_bfloat16* __restrict__ Xbf) {
  int idx = blockIdx.x * blockDim.x + threadIdx.x;
  if (idx >= B_ * T_ * F_) return;
  int e   = idx & 31;
  int row = idx >> 5;            // t*128 + b
  int b   = row & (B_ - 1);
  int t   = row >> 7;
  float v = 0.0f;
  if (e < E_) {
    float acc = b_emb[e];
    const float* dp = data + ((size_t)b * T_ + t) * F_;
#pragma unroll
    for (int f = 0; f < F_; ++f) acc += dp[f] * W_emb[f * E_ + e];
    v = sigmoidf_(acc);
  }
  Xbf[idx] = __float2bfloat16(v);
}

// ---------------- device-wide barrier (persistent kernel, NBLK resident blocks) ----------------
static __device__ __forceinline__ void grid_barrier(unsigned* bar, unsigned target) {
  __threadfence();
  __syncthreads();
  if (threadIdx.x == 0) {
    __hip_atomic_fetch_add(bar, 1u, __ATOMIC_ACQ_REL, __HIP_MEMORY_SCOPE_AGENT);
    while (__hip_atomic_load(bar, __ATOMIC_ACQUIRE, __HIP_MEMORY_SCOPE_AGENT) < target) {
      __builtin_amdgcn_s_sleep(4);
    }
  }
  __syncthreads();
}

// ---------------- persistent LSTM kernel ----------------
__global__ void __launch_bounds__(NTHR)
lstm_persistent(const __hip_bfloat16* __restrict__ Xbf,
                const __hip_bfloat16* __restrict__ WhT,
                const __hip_bfloat16* __restrict__ WxT,
                const float* __restrict__ b_lstm,
                const float* __restrict__ W_out,
                const float* __restrict__ b_out,
                float* __restrict__ gates,   // [128][2048]
                float* __restrict__ cbuf,    // [128][512]
                float* __restrict__ hf,      // [128][512] f32 (for output head)
                __hip_bfloat16* __restrict__ hbf, // [128][512] bf16 (next-step A)
                unsigned* __restrict__ bar,
                float* __restrict__ out) {   // [128][1024]
  // 64 local columns of WhT, each 512 bf16 (1024 B), padded to 1040 B rows.
  __shared__ __align__(16) unsigned char ldsWh[64 * LDS_ROW_B];

  const int wave = (int)blockIdx.x * (NTHR / 32) + ((int)threadIdx.x >> 5); // 0..255
  const int lane = (int)threadIdx.x & 31;
  const int nloc = lane & 15;   // M (for A) / N (for B,C,D) within tile
  const int half = lane >> 4;   // selects K-octet (A/B) or row-half (C/D)
  const int tile_m = wave & 7;          // 8 row-tiles of gates
  const int tnb    = (int)blockIdx.x * 4;  // 4 consecutive col-tiles, same for all waves in block
  const int arow   = tile_m * 16 + nloc;   // batch row this lane loads for A
  const int tid    = (int)blockIdx.x * NTHR + (int)threadIdx.x;
  const float bout0 = b_out[0];

  // ---- Stage this block's 64-column Wh slice into LDS (loop-invariant over t) ----
  {
    const int r = (int)threadIdx.x >> 2;   // 0..63 local column
    const int q = (int)threadIdx.x & 3;    // quarter of the 64-U4 row
    const U4* src = (const U4*)(WhT + ((size_t)blockIdx.x * 64 + r) * U_);
    U4* dst = (U4*)(ldsWh + (size_t)r * LDS_ROW_B);
#pragma unroll
    for (int i = 0; i < 16; ++i) dst[q * 16 + i] = src[q * 16 + i];
  }
  __syncthreads();

  float bias[4];
  const U4* Bx[4];
  const U4* Bl[4];   // LDS B-tile base per col-tile
#pragma unroll
  for (int j = 0; j < 4; ++j) {
    const int col = (tnb + j) * 16 + nloc;
    bias[j] = b_lstm[col];
    Bx[j] = (const U4*)(WxT + (size_t)col * F_);                    // 4 U4 per row
    Bl[j] = (const U4*)(ldsWh + (size_t)(j * 16 + nloc) * LDS_ROW_B); // 65 U4-pitch rows
  }
  const U4* Aw = (const U4*)(hbf + (size_t)arow * U_);

  unsigned barcnt = 0;

  for (int t = 0; t < T_; ++t) {
    // ---- Phase A: gates = x_t @ Wx + h @ Wh + b (bf16 WMMA, f32 accumulate) ----
    v8f z = {0.f,0.f,0.f,0.f,0.f,0.f,0.f,0.f};
    v8f acc[4] = {z, z, z, z};

    // x_t @ Wx : single K=32 chunk (padded from E=24)
    {
      const U4* Ax = (const U4*)(Xbf + ((size_t)t * B_ + arow) * F_);
      if (t + 1 < T_)  // prefetch next timestep's A tile (global_prefetch_b8)
        __builtin_prefetch(Xbf + ((size_t)(t + 1) * B_ + arow) * F_, 0, 0);
      U8 ra; ra.lo = Ax[half]; ra.hi = Ax[2 + half];
      v16bf a = __builtin_bit_cast(v16bf, ra);
#pragma unroll
      for (int j = 0; j < 4; ++j) {
        U8 rb; rb.lo = Bx[j][half]; rb.hi = Bx[j][2 + half];
        acc[j] = __builtin_amdgcn_wmma_f32_16x16x32_bf16(
            false, a, false, __builtin_bit_cast(v16bf, rb), (short)0, acc[j], false, false);
      }
    }

    // h @ Wh : K = 512 in 16 chunks of 32; A from global (fresh h), B from LDS
    for (int kc = 0; kc < 16; ++kc) {
      U8 ra; ra.lo = Aw[kc * 4 + half]; ra.hi = Aw[kc * 4 + 2 + half];
      v16bf a = __builtin_bit_cast(v16bf, ra);
#pragma unroll
      for (int j = 0; j < 4; ++j) {
        U8 rb; rb.lo = Bl[j][kc * 4 + half]; rb.hi = Bl[j][kc * 4 + 2 + half];
        acc[j] = __builtin_amdgcn_wmma_f32_16x16x32_bf16(
            false, a, false, __builtin_bit_cast(v16bf, rb), (short)0, acc[j], false, false);
      }
    }

    // store gates tile (+bias): lane holds col N=nloc, rows r+8*half per the C/D layout
#pragma unroll
    for (int j = 0; j < 4; ++j) {
      const int col = (tnb + j) * 16 + nloc;
      float* gp = gates + (size_t)(tile_m * 16 + half * 8) * G4_ + col;
#pragma unroll
      for (int r = 0; r < 8; ++r) gp[(size_t)r * G4_] = acc[j][r] + bias[j];
    }

    grid_barrier(bar, ++barcnt * NBLK);

    // ---- Phase B: elementwise LSTM cell update over [128,512] ----
    for (int e = tid; e < B_ * U_; e += NBLK * NTHR) {
      int b = e >> 9, u = e & (U_ - 1);
      const float* g = gates + (size_t)b * G4_;
      float ig = sigmoidf_(g[u]);
      float fg = sigmoidf_(g[U_ + u]);
      float og = sigmoidf_(g[2 * U_ + u]);
      float gg = tanhf(g[3 * U_ + u]);
      float c  = fg * cbuf[e] + ig * gg;
      float h  = og * tanhf(c);
      cbuf[e] = c;
      hf[e]   = h;
      hbf[e]  = __float2bfloat16(h);
    }

    grid_barrier(bar, ++barcnt * NBLK);

    // ---- Phase C: y[b,t] = sigmoid(h . W_out + b_out); overlaps next Phase A ----
    if (wave < B_) {
      float p = 0.0f;
      const float* hp = hf + (size_t)wave * U_;
#pragma unroll
      for (int j = 0; j < U_ / 32; ++j) p += hp[lane + 32 * j] * W_out[lane + 32 * j];
#pragma unroll
      for (int off = 16; off > 0; off >>= 1) p += __shfl_xor(p, off, 32);
      if (lane == 0) out[(size_t)wave * T_ + t] = sigmoidf_(p + bout0);
    }
  }
}

// ---------------- host launcher ----------------
extern "C" void kernel_launch(void* const* d_in, const int* in_sizes, int n_in,
                              void* d_out, int out_size, void* d_ws, size_t ws_size,
                              hipStream_t stream) {
  const float* data   = (const float*)d_in[0]; // [128,1024,32]
  const float* W_emb  = (const float*)d_in[1]; // [32,24]
  const float* b_emb  = (const float*)d_in[2]; // [24]
  const float* Wx     = (const float*)d_in[3]; // [24,2048]
  const float* Wh     = (const float*)d_in[4]; // [512,2048]
  const float* b_lstm = (const float*)d_in[5]; // [2048]
  const float* W_out  = (const float*)d_in[6]; // [512,1]
  const float* b_out  = (const float*)d_in[7]; // [1]
  float* out = (float*)d_out;                  // [128,1024,1]

  char* p = (char*)d_ws;
  __hip_bfloat16* WhT  = (__hip_bfloat16*)(p);             // 2,097,152 B
  __hip_bfloat16* WxT  = (__hip_bfloat16*)(p +  2097152);  //   131,072 B
  __hip_bfloat16* Xbf  = (__hip_bfloat16*)(p +  2228224);  // 8,388,608 B
  float*          gts  = (float*)         (p + 10616832);  // 1,048,576 B
  float*          cbuf = (float*)         (p + 11665408);  //   262,144 B
  float*          hfb  = (float*)         (p + 11927552);  //   262,144 B
  __hip_bfloat16* hbf  = (__hip_bfloat16*)(p + 12189696);  //   131,072 B
  unsigned*       bar  = (unsigned*)      (p + 12320768);  //       256 B

  // zero h0, c0, hf, and the barrier counter (graph-capture-safe, runs each replay)
  hipMemsetAsync(p + 11665408, 0, 655616, stream);

  // weight transpose/convert: 2048*512 + 2048*32 = 1,114,112 elements
  prep_kernel<<<(1114112 + NTHR - 1) / NTHR, NTHR, 0, stream>>>(Wh, Wx, WhT, WxT);

  // embedding: 128*1024*32 = 4,194,304 elements
  embed_kernel<<<(B_ * T_ * F_) / NTHR, NTHR, 0, stream>>>(data, W_emb, b_emb, Xbf);

  // persistent LSTM over all 1024 timesteps
  lstm_persistent<<<NBLK, NTHR, 0, stream>>>(Xbf, WhT, WxT, b_lstm, W_out, b_out,
                                             gts, cbuf, hfb, hbf, bar, out);
}
